// SPDToVector_84095459656161
// MI455X (gfx1250) — matrix operand order repaired and checked by reading, not compiled, not run on hardware
//
#include <hip/hip_runtime.h>
#include <hip/hip_bf16.h>

// ---------------------------------------------------------------------------
// SPD matrix-log (inverse scaling & squaring + Newton-Schulz sqrt, all WMMA)
// followed by a f16-WMMA projection GEMM.
//   S: (64,512,32,32) f32   W: (256,528) f32   b: (256) f32
//   out: (64,512,256) f32
// Workspace: v (32768 x 544 f16)  |  W16 (256 x 544 f16)
// ---------------------------------------------------------------------------

typedef __attribute__((ext_vector_type(2)))  float     v2f;
typedef __attribute__((ext_vector_type(8)))  float     v8f;
typedef __attribute__((ext_vector_type(4)))  _Float16  v4h;
typedef __attribute__((ext_vector_type(8)))  _Float16  v8h;
typedef __attribute__((ext_vector_type(16))) _Float16  v16h;

#define N_MAT     32768      // 64*512
#define S_DIM     32
#define DIM_VEC   528        // 32*33/2
#define VPAD      544        // 528 padded to multiple of 32 (f16 WMMA K)
#define OUT_DIM   256
#define LDSTRIDE  36         // 32x32 tile row stride: conflict-free AND 16B-aligned
#define MATSZ     (S_DIM * LDSTRIDE)
#define NS_ITERS  5
#define SQRT_LVLS 4          // T = S'^(1/16)

// C(32x32) = A(32x32) @ B(32x32), all in LDS with LDSTRIDE rows.
// mode 0: C = A@B        mode 1: C = 1.5*I - 0.5*(A@B)   (Newton-Schulz P)
// 4 waves: wave w owns output tile (w>>1, w&1); 8 k-steps of 16x16x4 f32 WMMA.
__device__ __forceinline__ void mm32(float* __restrict__ C,
                                     const float* __restrict__ A,
                                     const float* __restrict__ B,
                                     int wave, int lane, int mode)
{
    const int ti = (wave >> 1) & 1;
    const int tj = wave & 1;
    const int m  = lane & 15;           // row (A) / col (B) owned by this lane
    const int kk = (lane >> 4) << 1;    // lanes 0-15: K {0,1}; 16-31: K {2,3}

    v8f acc = {0.f, 0.f, 0.f, 0.f, 0.f, 0.f, 0.f, 0.f};
#pragma unroll
    for (int kb = 0; kb < 8; ++kb) {
        const int k0 = kb * 4 + kk;
        v2f a, b;
        a.x = A[(ti * 16 + m) * LDSTRIDE + k0];
        a.y = A[(ti * 16 + m) * LDSTRIDE + k0 + 1];
        b.x = B[(k0)     * LDSTRIDE + tj * 16 + m];
        b.y = B[(k0 + 1) * LDSTRIDE + tj * 16 + m];
        acc = __builtin_amdgcn_wmma_f32_16x16x4_f32(
            false, a, false, b, (short)0, acc, false, false);
    }

    const int mbase = ti * 16 + ((lane >> 4) << 3);  // VGPR i -> M = i (+8 hi half)
#pragma unroll
    for (int i = 0; i < 8; ++i) {
        const int row = mbase + i;
        const int col = tj * 16 + m;
        float v = acc[i];
        if (mode == 1)
            v = (row == col ? 1.5f : 0.0f) - 0.5f * v;
        C[row * LDSTRIDE + col] = v;
    }
}

// One block (128 threads = 4 wave32) per 32x32 SPD matrix.
__global__ __launch_bounds__(128)
void spd_logm_kernel(const float* __restrict__ S, _Float16* __restrict__ v_out)
{
    __shared__ __align__(16) float lds[5 * MATSZ];
    __shared__ float red[128];
    __shared__ float s_logfro, s_scale;

    const int n    = blockIdx.x;
    const int tid  = threadIdx.x;
    const int wave = tid >> 5;
    const int lane = tid & 31;

    float* pY = lds + 0 * MATSZ;   // starts as scaled S, becomes S'^(1/2^L)
    float* pZ = lds + 1 * MATSZ;
    float* pP = lds + 2 * MATSZ;
    float* pU = lds + 3 * MATSZ;
    float* pV = lds + 4 * MATSZ;

    // ---- load S (vectorized), accumulate squared Frobenius norm -------------
    // thread t owns 8 contiguous floats of one row: r = t>>2, c0 = (t&3)*8
    const float4* Sv = (const float4*)(S + (size_t)n * (S_DIM * S_DIM));
    const int r  = tid >> 2;
    const int c0 = (tid & 3) * 8;
    float4 x0 = Sv[tid * 2];
    float4 x1 = Sv[tid * 2 + 1];
    *(float4*)(&pY[r * LDSTRIDE + c0])     = x0;
    *(float4*)(&pY[r * LDSTRIDE + c0 + 4]) = x1;
    red[tid] = x0.x*x0.x + x0.y*x0.y + x0.z*x0.z + x0.w*x0.w
             + x1.x*x1.x + x1.y*x1.y + x1.z*x1.z + x1.w*x1.w;
    __syncthreads();
    for (int off = 64; off > 0; off >>= 1) {
        if (tid < off) red[tid] += red[tid + off];
        __syncthreads();
    }
    if (tid == 0) {
        const float fro = sqrtf(fmaxf(red[0], 1e-20f));
        s_scale  = 1.0f / fro;       // eigenvalues of S' = S/fro lie in (0,1]
        s_logfro = logf(fro);        // log(S) = log(S') + log(fro)*I
    }
    __syncthreads();

    {   // scale in place (vectorized)
        const float sc = s_scale;
        float4* p0 = (float4*)(&pY[r * LDSTRIDE + c0]);
        float4* p1 = (float4*)(&pY[r * LDSTRIDE + c0 + 4]);
        float4 a = *p0, b = *p1;
        a.x *= sc; a.y *= sc; a.z *= sc; a.w *= sc;
        b.x *= sc; b.y *= sc; b.z *= sc; b.w *= sc;
        *p0 = a; *p1 = b;
    }
    __syncthreads();

    // ---- 4 levels of matrix sqrt: coupled Newton-Schulz (inversion-free) ----
    // Y_{k+1} = Y_k P_k, Z_{k+1} = P_k Z_k, P_k = (3I - Z_k Y_k)/2
    for (int lvl = 0; lvl < SQRT_LVLS; ++lvl) {
        // Z = I
        for (int q = tid; q < S_DIM * S_DIM; q += 128) {
            const int rr = q >> 5, cc = q & 31;
            pZ[rr * LDSTRIDE + cc] = (rr == cc) ? 1.0f : 0.0f;
        }
        __syncthreads();

        for (int it = 0; it < NS_ITERS; ++it) {
            mm32(pP, pZ, pY, wave, lane, 1);            // P = 1.5I - 0.5*Z@Y
            __syncthreads();
            mm32(pU, pY, pP, wave, lane, 0);            // Ynew
            mm32(pV, pP, pZ, wave, lane, 0);            // Znew
            __syncthreads();
            float* t;
            t = pY; pY = pU; pU = t;
            t = pZ; pZ = pV; pV = t;
        }
        // pY now holds sqrt of previous T; next level consumes it directly.
    }

    // ---- log(T) ~ E - E^2/2 + E^3/3,  E = T - I;  T = S'^(1/16) -------------
    for (int q = tid; q < S_DIM * S_DIM; q += 128) {
        const int rr = q >> 5, cc = q & 31;
        pP[rr * LDSTRIDE + cc] = pY[rr * LDSTRIDE + cc] - (rr == cc ? 1.0f : 0.0f);
    }
    __syncthreads();
    mm32(pU, pP, pP, wave, lane, 0);                    // E^2
    __syncthreads();
    mm32(pV, pU, pP, wave, lane, 0);                    // E^3
    __syncthreads();

    // ---- extract lower triangle -> f16, zero-pad to VPAD --------------------
    // Packed 4-wide stores: threads 0..135 each own 4 consecutive indices.
    _Float16* vout = v_out + (size_t)n * VPAD;
    const float lf = s_logfro;
    if (tid < VPAD / 4) {
        v4h pack;
#pragma unroll
        for (int p = 0; p < 4; ++p) {
            const int idx = tid * 4 + p;
            float l = 0.f;
            if (idx < DIM_VEC) {
                int i = (int)((sqrtf(8.0f * (float)idx + 1.0f) - 1.0f) * 0.5f);
                while ((i + 1) * (i + 2) / 2 <= idx) ++i;
                while (i * (i + 1) / 2 > idx) --i;
                const int j   = idx - i * (i + 1) / 2;
                const int off = i * LDSTRIDE + j;
                l = 16.0f * (pP[off] - 0.5f * pU[off] + (1.0f / 3.0f) * pV[off]);
                if (i == j) l += lf;
            }
            pack[p] = (_Float16)l;
        }
        *(v4h*)(vout + tid * 4) = pack;
    }
}

// One-shot W (256x528 f32) -> W16 (256x544 f16, zero-padded) conversion.
__global__ __launch_bounds__(256)
void cvt_w_kernel(const float* __restrict__ W, _Float16* __restrict__ W16)
{
    const int idx = blockIdx.x * 256 + threadIdx.x;
    if (idx >= OUT_DIM * VPAD) return;
    const int col = idx / VPAD;
    const int k   = idx - col * VPAD;
    W16[idx] = (k < DIM_VEC) ? (_Float16)W[(size_t)col * DIM_VEC + k]
                             : (_Float16)0.0f;
}

// Projection GEMM: out(32768,256) = v(32768,544f16) @ W16^T(544,256) + b
// Block: 16 rows x 256 cols. 4 waves, 4 column-tiles each. K = 17 steps of 32.
__global__ __launch_bounds__(128)
void spd_proj_kernel(const _Float16* __restrict__ v_in,
                     const _Float16* __restrict__ W16,
                     const float* __restrict__ bias,
                     float* __restrict__ out)
{
    __shared__ __align__(16) _Float16 As[16 * VPAD];

    const int tid  = threadIdx.x;
    const int wave = tid >> 5;
    const int lane = tid & 31;
    const int row0 = blockIdx.x * 16;

    // Stage 16 rows of v into LDS with b128 copies (rows f16, padded, aligned).
    {
        const uint4* src = (const uint4*)(v_in + (size_t)row0 * VPAD);
        uint4*       dst = (uint4*)As;
        for (int q = tid; q < 16 * VPAD / 8; q += 128)
            dst[q] = src[q];
    }
    __syncthreads();

    const int m    = lane & 15;          // A row / B col owned by this lane
    const int half = lane >> 4;

    v8f acc[4] = {{0.f,0.f,0.f,0.f,0.f,0.f,0.f,0.f},
                  {0.f,0.f,0.f,0.f,0.f,0.f,0.f,0.f},
                  {0.f,0.f,0.f,0.f,0.f,0.f,0.f,0.f},
                  {0.f,0.f,0.f,0.f,0.f,0.f,0.f,0.f}};

    for (int k0 = 0; k0 < VPAD; k0 += 32) {
        // A fragment (16x32 f16): element e -> K = k0 + half*16 + e
        const v16h a = *(const v16h*)(&As[m * VPAD + k0 + half * 16]);

#pragma unroll
        for (int t = 0; t < 4; ++t) {
            const int col = (wave * 4 + t) * 16 + m;   // output column
            // B fragment (32x16): element e -> K = k0 + half*16 + e, col fixed.
            // 16 contiguous f16, 32B-aligned -> two global_load_b128.
            const v16h b = *(const v16h*)(W16 + (size_t)col * VPAD + k0 + half * 16);
            acc[t] = __builtin_amdgcn_wmma_f32_16x16x32_f16(
                false, a, false, b, (short)0, acc[t], false, false);
        }
    }

    const int mb = half << 3;  // VGPR i -> row M = mb + i
#pragma unroll
    for (int t = 0; t < 4; ++t) {
        const int col = (wave * 4 + t) * 16 + m;
        const float bc = bias[col];
#pragma unroll
        for (int i = 0; i < 8; ++i) {
            const int gr = row0 + mb + i;
            out[(size_t)gr * OUT_DIM + col] = acc[t][i] + bc;
        }
    }
}

extern "C" void kernel_launch(void* const* d_in, const int* in_sizes, int n_in,
                              void* d_out, int out_size, void* d_ws, size_t ws_size,
                              hipStream_t stream)
{
    (void)in_sizes; (void)n_in; (void)out_size; (void)ws_size;
    const float* S = (const float*)d_in[0];
    const float* W = (const float*)d_in[1];
    const float* b = (const float*)d_in[2];
    float*       out = (float*)d_out;

    _Float16* v   = (_Float16*)d_ws;                       // 32768 x 544 f16
    _Float16* W16 = v + (size_t)N_MAT * VPAD;              // 256 x 544 f16

    cvt_w_kernel<<<(OUT_DIM * VPAD + 255) / 256, 256, 0, stream>>>(W, W16);
    spd_logm_kernel<<<N_MAT, 128, 0, stream>>>(S, v);
    spd_proj_kernel<<<N_MAT / 16, 128, 0, stream>>>(v, W16, b, out);
}
